// SNNConvNet_29815662969400
// MI455X (gfx1250) — matrix operand order repaired and checked by reading, not compiled, not run on hardware
//
#include <hip/hip_runtime.h>
#include <stdint.h>
#include <stddef.h>

// ---------------------------------------------------------------------------
// SNN TTFS conv-net on gfx1250 (MI455X), wave32 + WMMA f32_16x16x32_f16.
// Time-serial LIF recurrence -> 35 steps x 4 fused kernels.
// All GEMMs (conv1/conv2 implicit-GEMM, fc1, fc2) run on v_wmma_f32_16x16x32_f16,
// with maxpool+LIF fused in-register via pool-group-ordered M dimension.
// conv2 stages its input image into LDS with GLOBAL_LOAD_ASYNC_TO_LDS_B128
// (ASYNCcnt path); fc1 streams K with global_prefetch_b8.
// ---------------------------------------------------------------------------

typedef __attribute__((ext_vector_type(16))) _Float16 v16h;
typedef __attribute__((ext_vector_type(8)))  float    v8f;

union V16 {
  v16h v;
  _Float16 e[16];
  unsigned int u[8];
};

// A-matrix (16x32 f16) K index for fragment element j (ISA 7.12.2):
// lane<16: j0..7 -> K0..7, j8..15 -> K16..23 ; lane>=16: K8..15 / K24..31
__device__ __forceinline__ int kOfA(int lane, int j) {
  return j + ((lane >= 16) ? 8 : 0) + ((j >= 8) ? 8 : 0);
}
// B-matrix (32x16 f16) K index (same as A): lane<16: K0..7,16..23 ; lane>=16: K8..15,24..31
__device__ __forceinline__ int kOfB(int lane, int j) {
  return j + ((lane >= 16) ? 8 : 0) + ((j >= 8) ? 8 : 0);
}

__device__ __forceinline__ v8f wmma_f16(const V16& a, const V16& b, v8f c) {
  // 8 args: (neg_a, A, neg_b, B, c_mod, C, reuse_a, reuse_b)
  return __builtin_amdgcn_wmma_f32_16x16x32_f16(false, a.v, false, b.v,
                                                (short)0, c, false, false);
}

#define BETA 0.95f
#define THR  0.9f

// ---------------------------------------------------------------------------
// Prep kernels
// ---------------------------------------------------------------------------

__global__ __launch_bounds__(256) void k_zero(unsigned int* __restrict__ p, int n) {
  int i = blockIdx.x * 256 + threadIdx.x;
  if (i < n) p[i] = 0u;
}

// TTFS latency encoding -> spike-time index map (u8), instead of [T,B,H,W] one-hot.
// t = clip(round(min(34*(1-x), 34*0.1)), 0, 34)
__global__ __launch_bounds__(256) void k_encode(const float* __restrict__ x,
                                                unsigned char* __restrict__ tidx, int n) {
  int i = blockIdx.x * 256 + threadIdx.x;
  if (i < n) {
    float v = fminf(34.0f * (1.0f - x[i]), 3.4f);
    v = rintf(v);
    v = fminf(fmaxf(v, 0.0f), 34.0f);
    tidx[i] = (unsigned char)v;
  }
}

// Convert + pad all weights to fp16 B-matrix-friendly layouts, pad biases.
// conv1: [16 n][32 k], conv2: [32 n][320 k] (k=ic*25+tap), fc1: [112][9152],
// fc2: [48][128], fc1 bias [112], fc2 bias [48].
__global__ __launch_bounds__(256) void k_prep(
    const float* __restrict__ c1w, const float* __restrict__ c2w,
    const float* __restrict__ f1w, const float* __restrict__ f2w,
    const float* __restrict__ f1b, const float* __restrict__ f2b,
    _Float16* __restrict__ c1wh, _Float16* __restrict__ c2wh,
    _Float16* __restrict__ f1wh, _Float16* __restrict__ f2wh,
    float* __restrict__ f1bp, float* __restrict__ f2bp) {
  int i = blockIdx.x * 256 + threadIdx.x;
  if (i < 512) {
    int n = i / 32, k = i % 32;
    c1wh[i] = (_Float16)((n < 12 && k < 25) ? c1w[n * 25 + k] : 0.0f);
  } else if (i < 10752) {
    int j = i - 512;
    int n = j / 320, k = j % 320;
    float v = 0.0f;
    if (k < 300) { int ic = k / 25, tap = k % 25; v = c2w[(n * 12 + ic) * 25 + tap]; }
    c2wh[j] = (_Float16)v;
  } else if (i < 1035776) {
    int j = i - 10752;
    int n = j / 9152, k = j % 9152;
    f1wh[j] = (_Float16)((n < 100) ? f1w[n * 9152 + k] : 0.0f);
  } else if (i < 1041920) {
    int j = i - 1035776;
    int n = j / 128, k = j % 128;
    f2wh[j] = (_Float16)((n < 35 && k < 100) ? f2w[n * 100 + k] : 0.0f);
  } else if (i < 1042032) {
    int j = i - 1041920;
    f1bp[j] = (j < 100) ? f1b[j] : 0.0f;
  } else if (i < 1042080) {
    int j = i - 1042032;
    f2bp[j] = (j < 35) ? f2b[j] : 0.0f;
  }
}

// ---------------------------------------------------------------------------
// conv1(5x5, 1->12) + maxpool2 + LIF, fused. Implicit GEMM:
// M = pool-group-ordered pre-pool pixels (15x48 pooled rows per half-image),
// N = 16 (12 real channels), K = 32 (25 real taps). One WMMA per tile.
// Pool window members are consecutive M rows -> in-register maxpool per lane.
// grid = 512 (b * 2 halves), 256 threads (8 waves).
// ---------------------------------------------------------------------------
__global__ __launch_bounds__(256) void k_conv1(
    const unsigned char* __restrict__ tidx,
    const _Float16* __restrict__ c1wh, const float* __restrict__ c1b,
    float* __restrict__ m1, _Float16* __restrict__ s1, int t) {
  const int b    = blockIdx.x >> 1;
  const int half = blockIdx.x & 1;
  const int lane = threadIdx.x & 31;
  const int wave = threadIdx.x >> 5;
  const int n    = lane & 15;

  V16 bf;
#pragma unroll
  for (int j = 0; j < 16; ++j) bf.e[j] = c1wh[n * 32 + kOfB(lane, j)];

  const unsigned char tt = (unsigned char)t;

  for (int tile = wave; tile < 180; tile += 8) {   // 720 groups / 4 per tile
    const int m = tile * 16 + (lane & 15);
    const int g = m >> 2, q = m & 3;               // pool group, quadrant
    const int y = half * 30 + 2 * (g / 48) + (q >> 1);  // pre-pool abs y (<60)
    const int x = 2 * (g % 48) + (q & 1);               // pre-pool x (<96)

    V16 af;
#pragma unroll
    for (int j = 0; j < 16; ++j) {
      const int k  = kOfA(lane, j);
      const int ky = k / 5, kx = k % 5;
      const bool ok = (k < 25);
      const int ai = ok ? ((b * 64 + (y + ky)) * 100 + (x + kx)) : 0;
      const unsigned char tv = tidx[ai];
      af.e[j] = (ok && (tv == tt)) ? (_Float16)1.0f : (_Float16)0.0f;
    }

    v8f acc = {};
    acc = wmma_f16(af, bf, acc);

    if (n < 12) {
      const float bias = c1b[n];
#pragma unroll
      for (int s = 0; s < 2; ++s) {
        const int gl = tile * 4 + ((lane >= 16) ? 2 : 0) + s;  // group this lane owns
        float v = fmaxf(fmaxf(acc[4 * s + 0], acc[4 * s + 1]),
                        fmaxf(acc[4 * s + 2], acc[4 * s + 3])) + bias;
        const int py = half * 15 + gl / 48;
        const int px = gl % 48;
        const size_t idx = ((size_t)(b * 12 + n) * 30 + py) * 48 + px;
        float mem = m1[idx];
        const float sp = (float)s1[idx];              // previous-step spike
        mem = BETA * mem + v - sp * THR;
        const float sn = (mem - THR) > 0.0f ? 1.0f : 0.0f;
        m1[idx] = mem;
        s1[idx] = (_Float16)sn;
      }
    }
  }
}

// ---------------------------------------------------------------------------
// conv2(5x5, 12->32) + maxpool2 + LIF, fused. Input image (34.5KB) staged in
// LDS via GLOBAL_LOAD_ASYNC_TO_LDS_B128 (direct global->LDS, ASYNCcnt).
// M = pool-group order over 26x44 (1144 rows, 72 tiles), N = 2x16 channels,
// K = 320 (300 real, 10 WMMA k-steps). grid = 256 (one image/WG), 256 threads.
// ---------------------------------------------------------------------------
__global__ __launch_bounds__(256) void k_conv2(
    const _Float16* __restrict__ s1,
    const _Float16* __restrict__ c2wh, const float* __restrict__ c2b,
    float* __restrict__ m2, _Float16* __restrict__ s2) {
  __shared__ _Float16 s1sh[12 * 30 * 48];          // 34,560 bytes
  const int b   = blockIdx.x;
  const int tid = threadIdx.x;
  {
    // Async DMA global->LDS, 16B per lane per issue, no VGPR staging.
    // LDS dest address = low 32 bits of the generic pointer (ISA: LDS aperture
    // maps addr[31:0] to the LDS offset).
    const uint8_t* gsrc = (const uint8_t*)(s1 + (size_t)b * 17280);
    const unsigned lds_base = (unsigned)(uintptr_t)(&s1sh[0]);
    for (int i = tid; i < 2160; i += 256) {        // 2160 * 16B = 34,560B
      const unsigned lds_addr = lds_base + (unsigned)(i * 16);
      const uint8_t* g = gsrc + (size_t)i * 16;
      asm volatile("global_load_async_to_lds_b128 %0, %1, off"
                   :: "v"(lds_addr), "v"(g)
                   : "memory");
    }
    asm volatile("s_wait_asynccnt 0" ::: "memory");
  }
  __syncthreads();

  const int lane = tid & 31;
  const int wave = tid >> 5;

  for (int wt = wave; wt < 144; wt += 8) {         // 72 M-tiles x 2 N-tiles
    const int tile = wt >> 1;
    const int nn   = (wt & 1) * 16 + (lane & 15);  // output channel 0..31
    const int m    = tile * 16 + (lane & 15);
    const int g = m >> 2, q = m & 3;
    const bool mok = (g < 286);                     // 13*22 pool groups
    const int y = 2 * (g / 22) + (q >> 1);
    const int x = 2 * (g % 22) + (q & 1);

    v8f acc = {};
    for (int kk = 0; kk < 10; ++kk) {
      V16 af, bf;
#pragma unroll
      for (int j = 0; j < 16; ++j) {
        const int kB = kk * 32 + kOfB(lane, j);
        bf.e[j] = c2wh[nn * 320 + kB];
        const int kA = kk * 32 + kOfA(lane, j);
        const int ic = kA / 25, tap = kA % 25;
        const int ky = tap / 5, kx = tap % 5;
        const bool ok = mok && (kA < 300);
        const int ai = ok ? ((ic * 30 + (y + ky)) * 48 + (x + kx)) : 0;
        const _Float16 av = s1sh[ai];
        af.e[j] = ok ? av : (_Float16)0.0f;
      }
      acc = wmma_f16(af, bf, acc);
    }

    const float bias = c2b[nn];
#pragma unroll
    for (int s = 0; s < 2; ++s) {
      const int gl = tile * 4 + ((lane >= 16) ? 2 : 0) + s;
      if (gl < 286) {
        float v = fmaxf(fmaxf(acc[4 * s + 0], acc[4 * s + 1]),
                        fmaxf(acc[4 * s + 2], acc[4 * s + 3])) + bias;
        const size_t idx = ((size_t)(b * 32 + nn) * 13 + gl / 22) * 22 + (gl % 22);
        float mem = m2[idx];
        const float sp = (float)s2[idx];
        mem = BETA * mem + v - sp * THR;
        const float sn = (mem - THR) > 0.0f ? 1.0f : 0.0f;
        m2[idx] = mem;
        s2[idx] = (_Float16)sn;                    // flat index == c*286+py*22+px
      }
    }
  }
}

// ---------------------------------------------------------------------------
// fc1 (9152 -> 100, padded N=112) + LIF. GEMM M=256, N=112 (7 tiles), K=9152
// (286 k-steps). 14 WG x 8 waves = 112 wave-tiles, one full-K tile per wave.
// A/B fragment K-pairs are memory-consecutive -> b32 loads per fragment dword.
// Streams K with global_prefetch_b8 ~512B ahead on both operands.
// ---------------------------------------------------------------------------
__global__ __launch_bounds__(256) void k_fc1(
    const _Float16* __restrict__ s2,
    const _Float16* __restrict__ f1wh, const float* __restrict__ f1bp,
    float* __restrict__ m3, _Float16* __restrict__ s3) {
  const int lane = threadIdx.x & 31;
  const int wave = threadIdx.x >> 5;
  const int wt = blockIdx.x * 8 + wave;            // 0..111
  const int mt = wt / 7, nt = wt % 7;
  const int row = mt * 16 + (lane & 15);           // batch row
  const int nn  = nt * 16 + (lane & 15);           // neuron (<112, rows padded)

  const unsigned int* Ab = (const unsigned int*)(s2 + (size_t)row * 9152);
  const unsigned int* Bb = (const unsigned int*)(f1wh + (size_t)nn * 9152);
  const int ah = (lane >= 16) ? 4 : 0;             // A pair-offset per lane half
  const int bh = (lane >= 16) ? 4 : 0;             // B pair-offset per lane half

  v8f acc = {};
  for (int kk = 0; kk < 286; ++kk) {
    const int pb = kk * 16;
    // Prefetch ~8 k-steps (512B) ahead; speculative, drop-on-fault, no counter.
    __builtin_prefetch((const char*)(Ab + pb) + 512, 0, 0);
    __builtin_prefetch((const char*)(Bb + pb) + 512, 0, 0);
    V16 af, bf;
#pragma unroll
    for (int v = 0; v < 8; ++v) {
      af.u[v] = Ab[pb + ah + v + ((v >= 4) ? 4 : 0)];
      bf.u[v] = Bb[pb + bh + v + ((v >= 4) ? 4 : 0)];
    }
    acc = wmma_f16(af, bf, acc);
  }

  const float bias = f1bp[nn];
#pragma unroll
  for (int r = 0; r < 8; ++r) {
    const int mm = mt * 16 + r + ((lane >= 16) ? 8 : 0);
    const size_t i3 = (size_t)mm * 112 + nn;
    const size_t is = (size_t)mm * 128 + nn;       // s3 padded to K=128 for fc2
    float mem = m3[i3];
    const float sp = (float)s3[is];
    const float cur = acc[r] + bias;
    mem = BETA * mem + cur - sp * THR;
    const float sn = (mem - THR) > 0.0f ? 1.0f : 0.0f;
    m3[i3] = mem;
    s3[is] = (_Float16)sn;
  }
}

// ---------------------------------------------------------------------------
// fc2 (100 -> 35, padded N=48, K=128) + LIF + output. 6 WG x 8 waves = 48 tiles.
// ---------------------------------------------------------------------------
__global__ __launch_bounds__(256) void k_fc2(
    const _Float16* __restrict__ s3,
    const _Float16* __restrict__ f2wh, const float* __restrict__ f2bp,
    float* __restrict__ m4, _Float16* __restrict__ s4,
    float* __restrict__ out, int t) {
  const int lane = threadIdx.x & 31;
  const int wave = threadIdx.x >> 5;
  const int wt = blockIdx.x * 8 + wave;            // 0..47
  const int mt = wt / 3, nt = wt % 3;
  const int row = mt * 16 + (lane & 15);
  const int nn  = nt * 16 + (lane & 15);           // <48 (weights zero-padded)

  const unsigned int* Ab = (const unsigned int*)(s3 + (size_t)row * 128);
  const unsigned int* Bb = (const unsigned int*)(f2wh + (size_t)nn * 128);
  const int ah = (lane >= 16) ? 4 : 0;
  const int bh = (lane >= 16) ? 4 : 0;

  v8f acc = {};
#pragma unroll
  for (int kk = 0; kk < 4; ++kk) {
    const int pb = kk * 16;
    V16 af, bf;
#pragma unroll
    for (int v = 0; v < 8; ++v) {
      af.u[v] = Ab[pb + ah + v + ((v >= 4) ? 4 : 0)];
      bf.u[v] = Bb[pb + bh + v + ((v >= 4) ? 4 : 0)];
    }
    acc = wmma_f16(af, bf, acc);
  }

  const float bias = f2bp[nn];
#pragma unroll
  for (int r = 0; r < 8; ++r) {
    const int mm = mt * 16 + r + ((lane >= 16) ? 8 : 0);
    const size_t i4 = (size_t)mm * 48 + nn;
    float mem = m4[i4];
    const float sp = (float)s4[i4];
    const float cur = acc[r] + bias;
    mem = BETA * mem + cur - sp * THR;
    const float sn = (mem - THR) > 0.0f ? 1.0f : 0.0f;
    m4[i4] = mem;
    s4[i4] = (_Float16)sn;
    if (nn < 35) out[((size_t)t * 256 + mm) * 35 + nn] = sn;
  }
}

// ---------------------------------------------------------------------------
// Host launcher (graph-capture safe, deterministic, ~45 MB of d_ws used)
// ---------------------------------------------------------------------------
extern "C" void kernel_launch(void* const* d_in, const int* in_sizes, int n_in,
                              void* d_out, int out_size, void* d_ws, size_t ws_size,
                              hipStream_t stream) {
  (void)in_sizes; (void)n_in; (void)out_size; (void)ws_size;
  const float* x   = (const float*)d_in[0];
  const float* c1w = (const float*)d_in[1];
  const float* c1b = (const float*)d_in[2];
  const float* c2w = (const float*)d_in[3];
  const float* c2b = (const float*)d_in[4];
  const float* f1w = (const float*)d_in[5];
  const float* f1b = (const float*)d_in[6];
  const float* f2w = (const float*)d_in[7];
  const float* f2b = (const float*)d_in[8];
  float* out = (float*)d_out;

  uint8_t* wsb = (uint8_t*)d_ws;
  size_t o = 0;
  auto take = [&](size_t bytes) -> void* {
    void* p = wsb + o;
    o += (bytes + 255) & ~(size_t)255;
    return p;
  };

  unsigned char* tidx = (unsigned char*)take((size_t)256 * 64 * 100);       // 1.6MB
  float*    m1 = (float*)   take((size_t)256 * 12 * 30 * 48 * 4);           // 17.7MB
  _Float16* s1 = (_Float16*)take((size_t)256 * 12 * 30 * 48 * 2);           // 8.8MB
  float*    m2 = (float*)   take((size_t)256 * 32 * 13 * 22 * 4);           // 9.4MB
  _Float16* s2 = (_Float16*)take((size_t)256 * 32 * 13 * 22 * 2);           // 4.7MB
  float*    m3 = (float*)   take((size_t)256 * 112 * 4);
  _Float16* s3 = (_Float16*)take((size_t)256 * 128 * 2);
  float*    m4 = (float*)   take((size_t)256 * 48 * 4);
  _Float16* s4 = (_Float16*)take((size_t)256 * 48 * 2);
  _Float16* c1wh = (_Float16*)take((size_t)16 * 32 * 2);
  _Float16* c2wh = (_Float16*)take((size_t)32 * 320 * 2);
  _Float16* f1wh = (_Float16*)take((size_t)112 * 9152 * 2);                 // 2.05MB
  _Float16* f2wh = (_Float16*)take((size_t)48 * 128 * 2);
  float* f1bp = (float*)take((size_t)112 * 4);
  float* f2bp = (float*)take((size_t)48 * 4);

  // Zero all persistent LIF state (m1..s4, incl. alignment pads) every call.
  const int zero_dwords = (int)(((uint8_t*)c1wh - (uint8_t*)m1) / 4);
  k_zero<<<(zero_dwords + 255) / 256, 256, 0, stream>>>((unsigned int*)m1, zero_dwords);

  k_encode<<<(1638400 + 255) / 256, 256, 0, stream>>>(x, tidx, 1638400);
  k_prep<<<(1042080 + 255) / 256, 256, 0, stream>>>(c1w, c2w, f1w, f2w, f1b, f2b,
                                                    c1wh, c2wh, f1wh, f2wh, f1bp, f2bp);

  for (int t = 0; t < 35; ++t) {
    k_conv1<<<512, 256, 0, stream>>>(tidx, c1wh, c1b, m1, s1, t);
    k_conv2<<<256, 256, 0, stream>>>(s1, c2wh, c2b, m2, s2);
    k_fc1  <<< 14, 256, 0, stream>>>(s2, f1wh, f1bp, m3, s3);
    k_fc2  <<<  6, 256, 0, stream>>>(s3, f2wh, f2bp, m4, s4, out, t);
  }
}
